// NERF_91259465105881
// MI455X (gfx1250) — matrix operand order repaired and checked by reading, not compiled, not run on hardware
//
#include <hip/hip_runtime.h>
#include <hip/hip_bf16.h>

typedef __attribute__((ext_vector_type(16))) _Float16 v16h;
typedef __attribute__((ext_vector_type(8)))  _Float16 v8h;
typedef __attribute__((ext_vector_type(8)))  float    v8f;

#define NEAR_P 0.1f
#define FAR_P  4.0f

__device__ __forceinline__ v8f wmma_f16(v16h a, v16h b, v8f c) {
  // D = A(16x32 f16) * B(32x16 f16) + C(16x16 f32)
  return __builtin_amdgcn_wmma_f32_16x16x32_f16(false, a, false, b, (short)0, c,
                                                false, false);
}

// Build a WMMA A-fragment from an LDS row-major activation buffer.
// ISA 16-bit A 16x32 layout: lane L holds row M=L%16;
// elements e=0..7  -> K = kbase + (L>=16 ? 8 : 0) + e
// elements e=8..15 -> K = kbase + 16 + (L>=16 ? 8 : 0) + (e-8)
__device__ __forceinline__ v16h ld_afrag(const _Float16* rowp, int kbase) {
  union { v16h v; v8h h[2]; } u;
  u.h[0] = *(const v8h*)(rowp + kbase);
  u.h[1] = *(const v8h*)(rowp + kbase + 16);
  return u.v;
}

// ---------------------------------------------------------------------------
// Weight packing: fp32 -> f16 in WMMA B-fragment order.
// B fragment (32x16, f16): lane L holds col N = L%16, K = (e/8)*16 + (L/16)*8 + e%8.
// Fragment index f = kc*CT + ct ; storage: [(f*32 + L)*16 + e]
// Offsets (halfs): w1 @0 (12288), w2 @12288 (16384), w3 @28672 (16384),
//                  w4 @45056 (2048, 128x16 zero-padded from 128x4)
// ---------------------------------------------------------------------------
__global__ __launch_bounds__(256) void pack_weights(
    const float* __restrict__ w_in, const float* __restrict__ w_h1,
    const float* __restrict__ w_h2, const float* __restrict__ w_out,
    _Float16* __restrict__ wp) {
  _Float16* w1 = wp;
  _Float16* w2 = wp + 12288;
  _Float16* w3 = wp + 28672;
  _Float16* w4 = wp + 45056;
  const int stride = gridDim.x * blockDim.x;
  const int t0 = blockIdx.x * blockDim.x + threadIdx.x;

  // Layer 1: K=96 (3 chunks), 8 col tiles
  for (int i = t0; i < 3 * 8 * 512; i += stride) {
    int e = i & 15, L = (i >> 4) & 31, f = i >> 9;
    int kc = f >> 3, ct = f & 7;
    int n = ct * 16 + (L & 15);
    int k = kc * 32 + ((e >> 3) << 4) + ((L >> 4) << 3) + (e & 7);
    w1[i] = (_Float16)w_in[k * 128 + n];
  }
  // Layers 2 & 3: K=128 (4 chunks), 8 col tiles
  for (int i = t0; i < 4 * 8 * 512; i += stride) {
    int e = i & 15, L = (i >> 4) & 31, f = i >> 9;
    int kc = f >> 3, ct = f & 7;
    int n = ct * 16 + (L & 15);
    int k = kc * 32 + ((e >> 3) << 4) + ((L >> 4) << 3) + (e & 7);
    w2[i] = (_Float16)w_h1[k * 128 + n];
    w3[i] = (_Float16)w_h2[k * 128 + n];
  }
  // Layer 4: K=128 (4 chunks), 1 col tile (cols 0..3 valid, rest zero)
  for (int i = t0; i < 4 * 512; i += stride) {
    int e = i & 15, L = (i >> 4) & 31, kc = i >> 9;
    int n = L & 15;
    int k = kc * 32 + ((e >> 3) << 4) + ((L >> 4) << 3) + (e & 7);
    w4[i] = (n < 4) ? (_Float16)w_out[k * 4 + n] : (_Float16)0.0f;
  }
}

// Hidden layer (128 -> 128) for TWO 16-row tiles owned by this wave: each
// B fragment load feeds two independent WMMA accumulator chains.
__device__ __forceinline__ void hid_layer128x2(
    _Float16 (*act)[136], int arow0, int arow1, int khi, int mrow0, int mrow1,
    int ncol, int lane, const v16h* __restrict__ wp,
    const float* __restrict__ bias) {
  v16h a0[4], a1[4];
#pragma unroll
  for (int kc = 0; kc < 4; ++kc) {
    a0[kc] = ld_afrag(&act[arow0][0], kc * 32 + khi);
    a1[kc] = ld_afrag(&act[arow1][0], kc * 32 + khi);
  }
#pragma unroll
  for (int ct = 0; ct < 8; ++ct) {
    v8f c0 = {};
    v8f c1 = {};
#pragma unroll
    for (int kc = 0; kc < 4; ++kc) {
      v16h b = wp[(kc * 8 + ct) * 32 + lane];
      c0 = wmma_f16(a0[kc], b, c0);
      c1 = wmma_f16(a1[kc], b, c1);
    }
    int n = ct * 16 + ncol;
    float bb = bias[n];
#pragma unroll
    for (int v = 0; v < 8; ++v) {
      float h0 = c0[v] + bb;
      float h1 = c1[v] + bb;
      act[mrow0 + v][n] = (_Float16)(h0 > 0.0f ? h0 : 0.0f);
      act[mrow1 + v][n] = (_Float16)(h1 > 0.0f ? h1 : 0.0f);
    }
  }
}

// ---------------------------------------------------------------------------
// Fused NeRF render pass. Block = 128 threads (4 wave32) = 128 sample points;
// each wave owns two 16-row tiles (B-fragment reuse x2).
// coarse: NPIX=4, SAMPLES=32 ; fine: NPIX=1, SAMPLES=128.
// ---------------------------------------------------------------------------
template <int SAMPLES, int NPIX, bool IS_FINE>
__global__ __launch_bounds__(128) void render_kernel(
    const float* __restrict__ pixel, const float* __restrict__ cam,
    const float* __restrict__ invK, const float* __restrict__ benc,
    const float* __restrict__ bias_in, const float* __restrict__ bias_h1,
    const float* __restrict__ bias_h2, const float* __restrict__ bias_out,
    const v16h* __restrict__ w1p, const v16h* __restrict__ w2p,
    const v16h* __restrict__ w3p, const v16h* __restrict__ w4p,
    const float* __restrict__ finez,     // [pix][128], fine only
    float* __restrict__ out_color, float* __restrict__ out_depth,
    float* __restrict__ out_var,
    float* __restrict__ cw_ws,           // [pix][32], coarse only
    float* __restrict__ out_fsdf,        // [s*B + pix], fine only
    int B) {
  constexpr int ROWS = SAMPLES * NPIX;   // 128
  __shared__ _Float16 act[ROWS][136];    // padded stride: 272B row, 16B-aligned
  __shared__ float predb[ROWS][4];
  __shared__ float zbuf[ROWS];
  __shared__ float wbuf[ROWS];
  __shared__ float bencs[144];
  __shared__ float rdt[NPIX][8];         // [0..2]=ray dir (R*invK*pix), [4..6]=t

  const int tid = threadIdx.x;
  const int lane = tid & 31;
  const int wave = tid >> 5;             // 0..3
  const int blk = blockIdx.x;

  if (tid < 128) {                       // benc is 144 floats: split the tail
    bencs[tid] = benc[tid];
    if (tid < 16) bencs[128 + tid] = benc[128 + tid];
  }
  if (tid < NPIX) {
    int gp = blk * NPIX + tid;
    float px = pixel[gp * 2 + 0], py = pixel[gp * 2 + 1];
    float d0 = invK[0] * px + invK[1] * py + invK[2];
    float d1 = invK[3] * px + invK[4] * py + invK[5];
    float d2 = invK[6] * px + invK[7] * py + invK[8];
    const float* cm = cam + (size_t)gp * 16;
#pragma unroll
    for (int i = 0; i < 3; ++i) {
      rdt[tid][i] = cm[i * 4 + 0] * d0 + cm[i * 4 + 1] * d1 + cm[i * 4 + 2] * d2;
      rdt[tid][4 + i] = cm[i * 4 + 3];
    }
  }
  // Warm WGP$/L2 with layer-1 weights while we wait at the barrier.
  __builtin_prefetch((const char*)w1p + (lane << 7), 0, 1);
  __syncthreads();

  // -------- positional encoding: one point per thread (all 128 threads) ----
  {
    int p = tid / SAMPLES;
    int s = tid % SAMPLES;
    float z;
    if (IS_FINE)
      z = finez[(size_t)(blk * NPIX + p) * 128 + s];
    else
      z = NEAR_P + (s + 0.5f) * ((FAR_P - NEAR_P) / (float)SAMPLES);
    zbuf[tid] = z;
    float ptx = z * rdt[p][0] + rdt[p][4];
    float pty = z * rdt[p][1] + rdt[p][5];
    float ptz = z * rdt[p][2] + rdt[p][6];
#pragma unroll 4
    for (int j = 0; j < 48; ++j) {
      float pr = ptx * bencs[j] + pty * bencs[48 + j] + ptz * bencs[96 + j];
      act[tid][j]      = (_Float16)__sinf(pr);
      act[tid][48 + j] = (_Float16)__cosf(pr);
    }
  }
  __syncthreads();

  // -------- MLP via WMMA: wave owns rows [wave*32, wave*32+32) -------------
  const int m0 = wave * 32;
  const int arow0 = m0 + (lane & 15);
  const int arow1 = arow0 + 16;
  const int khi = (lane >> 4) << 3;          // A K sub-offset (0 or 8)
  const int mrow0 = m0 + ((lane >> 4) << 3); // D row base, tile 0
  const int mrow1 = mrow0 + 16;              // D row base, tile 1
  const int ncol = lane & 15;                // D col within tile

  // Layer 1: enc(96) -> hid(128)
  __builtin_prefetch((const char*)w2p + (lane << 7), 0, 1);
  {
    v16h a0[3], a1[3];
#pragma unroll
    for (int kc = 0; kc < 3; ++kc) {
      a0[kc] = ld_afrag(&act[arow0][0], kc * 32 + khi);
      a1[kc] = ld_afrag(&act[arow1][0], kc * 32 + khi);
    }
#pragma unroll
    for (int ct = 0; ct < 8; ++ct) {
      v8f c0 = {};
      v8f c1 = {};
#pragma unroll
      for (int kc = 0; kc < 3; ++kc) {
        v16h b = w1p[(kc * 8 + ct) * 32 + lane];
        c0 = wmma_f16(a0[kc], b, c0);
        c1 = wmma_f16(a1[kc], b, c1);
      }
      int n = ct * 16 + ncol;
      float bb = bias_in[n];
#pragma unroll
      for (int v = 0; v < 8; ++v) {
        float h0 = c0[v] + bb;
        float h1 = c1[v] + bb;
        act[mrow0 + v][n] = (_Float16)(h0 > 0.0f ? h0 : 0.0f);
        act[mrow1 + v][n] = (_Float16)(h1 > 0.0f ? h1 : 0.0f);
      }
    }
  }
  // Layers 2 & 3: hid(128) -> hid(128)
  __builtin_prefetch((const char*)w3p + (lane << 7), 0, 1);
  hid_layer128x2(act, arow0, arow1, khi, mrow0, mrow1, ncol, lane, w2p, bias_h1);
  __builtin_prefetch((const char*)w4p + (lane << 7), 0, 1);
  hid_layer128x2(act, arow0, arow1, khi, mrow0, mrow1, ncol, lane, w3p, bias_h2);
  // Layer 4: hid(128) -> 4 (padded to 16 cols)
  {
    v16h a0[4], a1[4];
#pragma unroll
    for (int kc = 0; kc < 4; ++kc) {
      a0[kc] = ld_afrag(&act[arow0][0], kc * 32 + khi);
      a1[kc] = ld_afrag(&act[arow1][0], kc * 32 + khi);
    }
    v8f c0 = {};
    v8f c1 = {};
#pragma unroll
    for (int kc = 0; kc < 4; ++kc) {
      v16h b = w4p[kc * 32 + lane];
      c0 = wmma_f16(a0[kc], b, c0);
      c1 = wmma_f16(a1[kc], b, c1);
    }
    if (ncol < 4) {
      float bb = bias_out[ncol];
#pragma unroll
      for (int v = 0; v < 8; ++v) {
        predb[mrow0 + v][ncol] = c0[v] + bb;
        predb[mrow1 + v][ncol] = c1[v] + bb;
      }
    }
  }
  __syncthreads();

  // -------- alpha compositing (sequential cumprod, one thread per pixel) ---
  if (tid < NPIX) {
    int gp = blk * NPIX + tid;
    float T = 1.0f, c0 = 0.f, c1 = 0.f, c2 = 0.f, dep = 0.f;
    for (int s = 0; s < SAMPLES; ++s) {
      int r = tid * SAMPLES + s;
      float sg = predb[r][3];
      sg = sg > 0.0f ? sg : 0.0f;
      float z = zbuf[r];
      float dz = (s == SAMPLES - 1) ? 100.0f : (zbuf[r + 1] - z);
      float alpha = 1.0f - __expf(-sg * dz);
      float w = alpha * T;
      T *= (1.0f - alpha + 1e-10f);
      wbuf[r] = w;
      c0 += w * (1.0f / (1.0f + __expf(-predb[r][0])));
      c1 += w * (1.0f / (1.0f + __expf(-predb[r][1])));
      c2 += w * (1.0f / (1.0f + __expf(-predb[r][2])));
      dep += w * z;
      if constexpr (!IS_FINE) cw_ws[(size_t)gp * 32 + s] = w;
    }
    float var = 0.0f;
    for (int s = 0; s < SAMPLES; ++s) {
      int r = tid * SAMPLES + s;
      float d = zbuf[r] - dep;
      var += wbuf[r] * d * d;
    }
    out_color[gp * 3 + 0] = c0;
    out_color[gp * 3 + 1] = c1;
    out_color[gp * 3 + 2] = c2;
    out_depth[gp] = dep;
    out_var[gp] = var;
  }
  if constexpr (IS_FINE) {
    if (tid < SAMPLES) out_fsdf[(size_t)tid * B + blk] = predb[tid][3];
  }
}

// ---------------------------------------------------------------------------
// Inverse-CDF sampling + streaming merge-sort (sample_pdf + sort(concat)).
// One pixel per thread; u ascending => searchsorted is a monotone scan, and
// fine samples come out sorted, so merging with the sorted coarse bins is a
// two-pointer merge written straight to the outputs.
// ---------------------------------------------------------------------------
__global__ __launch_bounds__(256) void pdf_kernel(
    const float* __restrict__ cw, float* __restrict__ fz_ws,
    float* __restrict__ out_fz, int B) {
  __shared__ float cdfs[256][33];
  int gp = blockIdx.x * 256 + threadIdx.x;
  if (gp >= B) return;
  float* cdf = cdfs[threadIdx.x];
  float sum = 0.0f;
  for (int k = 0; k < 32; ++k) {
    float w = cw[(size_t)gp * 32 + k] + 1e-5f;
    sum += w;
    cdf[k] = sum;
  }
  float inv = 1.0f / sum;
  for (int k = 0; k < 32; ++k) cdf[k] *= inv;

  const float bw = (FAR_P - NEAR_P) / 32.0f;
  int sidx = 0;
  auto fine_sample = [&](int ii) -> float {
    float u = (ii + 0.5f) * (1.0f / 96.0f);
    while (sidx < 32 && cdf[sidx] < u) ++sidx;   // searchsorted(side='left')
    int below = sidx - 1;
    below = below < 0 ? 0 : below;
    int above = sidx > 31 ? 31 : sidx;
    float cb = cdf[below], ca = cdf[above];
    float denom = (ca - cb < 1e-5f) ? 1.0f : (ca - cb);
    float t = (u - cb) / denom;
    float bb = NEAR_P + (below + 0.5f) * bw;
    float ba = NEAR_P + (above + 0.5f) * bw;
    return bb + t * (ba - bb);
  };

  int i = 0, j = 0;
  float fv = fine_sample(0);
  float cv = NEAR_P + 0.5f * bw;
  for (int o = 0; o < 128; ++o) {
    float v;
    bool takeFine = (i < 96) && (j >= 32 || fv <= cv);
    if (takeFine) {
      v = fv;
      ++i;
      if (i < 96) fv = fine_sample(i);
    } else {
      v = cv;
      ++j;
      cv = NEAR_P + (j + 0.5f) * bw;
    }
    fz_ws[(size_t)gp * 128 + o] = v;       // pixel-major for fine pass
    out_fz[(size_t)o * B + gp] = v;        // sample-major output
  }
}

// ---------------------------------------------------------------------------
// Host launch. Outputs (flat, fp32): c_color 3b | c_depth b | c_var b |
// f_color 3b | f_depth b | f_var b | f_sdf 128b | fine_z 128b.
// Workspace: packed f16 weights (94208 B) | c_w b*32 f32 | fine_z b*128 f32.
// ---------------------------------------------------------------------------
extern "C" void kernel_launch(void* const* d_in, const int* in_sizes, int n_in,
                              void* d_out, int out_size, void* d_ws,
                              size_t ws_size, hipStream_t stream) {
  const float* pixel = (const float*)d_in[0];
  const float* cam   = (const float*)d_in[2];
  const float* invK  = (const float*)d_in[4];
  const float* benc  = (const float*)d_in[5];
  const float* w_in  = (const float*)d_in[6];
  const float* b_in  = (const float*)d_in[7];
  const float* w_h1  = (const float*)d_in[8];
  const float* b_h1  = (const float*)d_in[9];
  const float* w_h2  = (const float*)d_in[10];
  const float* b_h2  = (const float*)d_in[11];
  const float* w_out = (const float*)d_in[12];
  const float* b_out = (const float*)d_in[13];
  float* out = (float*)d_out;
  const int b = in_sizes[0] / 2;  // 4096 pixels

  _Float16* wpack = (_Float16*)d_ws;
  float* cw = (float*)((char*)d_ws + 94208);
  float* fz = (float*)((char*)d_ws + 94208 + (size_t)b * 32 * sizeof(float));
  const v16h* w1p = (const v16h*)(wpack);
  const v16h* w2p = (const v16h*)(wpack + 12288);
  const v16h* w3p = (const v16h*)(wpack + 28672);
  const v16h* w4p = (const v16h*)(wpack + 45056);

  pack_weights<<<64, 256, 0, stream>>>(w_in, w_h1, w_h2, w_out, wpack);

  render_kernel<32, 4, false><<<b / 4, 128, 0, stream>>>(
      pixel, cam, invK, benc, b_in, b_h1, b_h2, b_out, w1p, w2p, w3p, w4p,
      nullptr, out + 0, out + 3 * b, out + 4 * b, cw, nullptr, b);

  pdf_kernel<<<(b + 255) / 256, 256, 0, stream>>>(cw, fz, out + 138 * b, b);

  render_kernel<128, 1, true><<<b, 128, 0, stream>>>(
      pixel, cam, invK, benc, b_in, b_h1, b_h2, b_out, w1p, w2p, w3p, w4p, fz,
      out + 5 * b, out + 8 * b, out + 9 * b, nullptr, out + 10 * b, b);
}
